// Generator_84679575208274
// MI455X (gfx1250) — matrix operand order repaired and checked by reading, not compiled, not run on hardware
//
#include <hip/hip_runtime.h>
#include <math.h>

// ---------------------------------------------------------------------------
// Gumbel-softmax LSTM generator for MI455X (gfx1250), bf16 WMMA + fp32 accum.
// 2x2 register-blocked WMMA macro-tiles on the weight-heavy GEMMs.
// ---------------------------------------------------------------------------

typedef __bf16 bf16_t;
typedef __attribute__((ext_vector_type(16))) __bf16 v16bf;
typedef __attribute__((ext_vector_type(8)))  __bf16 v8bf;
typedef __attribute__((ext_vector_type(8)))  float  v8f;

namespace {
constexpr int kVocab = 5000;
constexpr int kVPad  = 5024;  // vocab padded to multiple of 32 (WMMA K for y@emb)
constexpr int kNPad  = 5056;  // vocab padded to multiple of 32 (N for logits, 316 tiles)
constexpr int kEmb   = 256;
constexpr int kHid   = 512;
constexpr int kLat   = 128;
constexpr int kBatch = 128;
constexpr int kTime  = 128;
constexpr int kGates = 2048;  // 4*kHid
}

// ---------------------------------------------------------------------------
// WMMA fragment loaders (wave32, v_wmma_f32_16x16x32_bf16 layouts).
//
// A (16x32 bf16, row-major src, ld = K stride in elements):
//   lanes 0-15 : M=lane,    elems 0..7 -> K=k0+0..7,  elems 8..15 -> K=k0+16..23
//   lanes 16-31: M=lane-16, elems 0..7 -> K=k0+8..15, elems 8..15 -> K=k0+24..31
__device__ __forceinline__ v16bf load_a_frag(const bf16_t* __restrict__ A, int ld,
                                             int m0, int k0, int lane) {
    int row = m0 + (lane & 15);
    const bf16_t* p = A + (size_t)row * ld + k0 + ((lane >> 4) & 1) * 8;
    v8bf lo = *(const v8bf*)(p);
    v8bf hi = *(const v8bf*)(p + 16);
    v16bf r;
#pragma unroll
    for (int i = 0; i < 8; ++i) { r[i] = lo[i]; r[i + 8] = hi[i]; }
    return r;
}

// B (32x16 bf16) loaded from BT[N][K] (i.e. column-major B => K contiguous):
//   lanes 0-15 : N=n0+lane,    elems 0..7 -> K=k0+0..7,  elems 8..15 -> K=k0+16..23
//   lanes 16-31: N=n0+lane-16, elems 0..7 -> K=k0+8..15, elems 8..15 -> K=k0+24..31
__device__ __forceinline__ v16bf load_b_frag(const bf16_t* __restrict__ BT, int ld,
                                             int n0, int k0, int lane) {
    int col = n0 + (lane & 15);
    const bf16_t* p = BT + (size_t)col * ld + k0 + ((lane >> 4) & 1) * 8;
    v8bf lo = *(const v8bf*)(p);
    v8bf hi = *(const v8bf*)(p + 16);
    v16bf r;
#pragma unroll
    for (int i = 0; i < 8; ++i) { r[i] = lo[i]; r[i + 8] = hi[i]; }
    return r;
}

#define WMMA_BF16(A, B, C) \
    __builtin_amdgcn_wmma_f32_16x16x32_bf16(false, (A), false, (B), (short)0, (C), false, false)

// D tile store: VGPR i -> row m0 + i + (lane>=16 ? 8 : 0), col n0 + (lane&15)
__device__ __forceinline__ void store_tile_f32(float* __restrict__ D, int ld,
                                               int m0, int n0, int lane, v8f acc) {
    int col   = n0 + (lane & 15);
    int rbase = m0 + ((lane >> 4) & 1) * 8;
#pragma unroll
    for (int i = 0; i < 8; ++i)
        D[(size_t)(rbase + i) * ld + col] = acc[i];
}

__device__ __forceinline__ float sigf(float x) { return 1.0f / (1.0f + __expf(-x)); }

// ---------------------------------------------------------------------------
// Per-step GEMM kernels
// ---------------------------------------------------------------------------

// x[128][256] = y_bf[128][5024] @ embT^T  (K = 5024 zero-padded; 1 tile/wave
// keeps 128 waves in flight across the long 157-iteration K loop)
__global__ void __launch_bounds__(256)
k_gemm_x(const bf16_t* __restrict__ ybf, const bf16_t* __restrict__ embT,
         bf16_t* __restrict__ xbf) {
    int wave = threadIdx.x >> 5, lane = threadIdx.x & 31;
    int tile = blockIdx.x * 8 + wave;     // 128 tiles = 8 (M) x 16 (N)
    int m0 = (tile >> 4) * 16;
    int n0 = (tile & 15) * 16;
    v8f acc = {};
    for (int k0 = 0; k0 < kVPad; k0 += 32) {
        v16bf a = load_a_frag(ybf, kVPad, m0, k0, lane);
        v16bf b = load_b_frag(embT, kVPad, n0, k0, lane);
        acc = WMMA_BF16(a, b, acc);
    }
    int col   = n0 + (lane & 15);
    int rbase = m0 + ((lane >> 4) & 1) * 8;
#pragma unroll
    for (int i = 0; i < 8; ++i)
        xbf[(size_t)(rbase + i) * kEmb + col] = (bf16_t)acc[i];
}

// gates[128][2048] = x_bf@W_ih^T + h_bf@W_hh^T + bias
// 2x2 macro-tile per wave: 256 groups = 4 (M) x 64 (N), 32 blocks x 8 waves.
__global__ void __launch_bounds__(256)
k_gemm_gates(const bf16_t* __restrict__ xbf, const bf16_t* __restrict__ hbf,
             const bf16_t* __restrict__ Wih, const bf16_t* __restrict__ Whh,
             const float* __restrict__ bias, float* __restrict__ gates) {
    int wave = threadIdx.x >> 5, lane = threadIdx.x & 31;
    int grp = blockIdx.x * 8 + wave;
    int m0 = (grp & 3) * 32;
    int n0 = (grp >> 2) * 32;
    int c0 = n0 + (lane & 15), c1 = c0 + 16;
    float bv0 = bias[c0], bv1 = bias[c1];
    v8f acc00, acc01, acc10, acc11;
#pragma unroll
    for (int i = 0; i < 8; ++i) {
        acc00[i] = bv0; acc01[i] = bv1; acc10[i] = bv0; acc11[i] = bv1;
    }
    for (int k0 = 0; k0 < kEmb; k0 += 32) {         // x @ W_ih^T
        v16bf a0 = load_a_frag(xbf, kEmb, m0,      k0, lane);
        v16bf a1 = load_a_frag(xbf, kEmb, m0 + 16, k0, lane);
        v16bf b0 = load_b_frag(Wih, kEmb, n0,      k0, lane);
        v16bf b1 = load_b_frag(Wih, kEmb, n0 + 16, k0, lane);
        acc00 = WMMA_BF16(a0, b0, acc00);
        acc01 = WMMA_BF16(a0, b1, acc01);
        acc10 = WMMA_BF16(a1, b0, acc10);
        acc11 = WMMA_BF16(a1, b1, acc11);
    }
    for (int k0 = 0; k0 < kHid; k0 += 32) {         // h @ W_hh^T
        v16bf a0 = load_a_frag(hbf, kHid, m0,      k0, lane);
        v16bf a1 = load_a_frag(hbf, kHid, m0 + 16, k0, lane);
        v16bf b0 = load_b_frag(Whh, kHid, n0,      k0, lane);
        v16bf b1 = load_b_frag(Whh, kHid, n0 + 16, k0, lane);
        acc00 = WMMA_BF16(a0, b0, acc00);
        acc01 = WMMA_BF16(a0, b1, acc01);
        acc10 = WMMA_BF16(a1, b0, acc10);
        acc11 = WMMA_BF16(a1, b1, acc11);
    }
    store_tile_f32(gates, kGates, m0,      n0,      lane, acc00);
    store_tile_f32(gates, kGates, m0,      n0 + 16, lane, acc01);
    store_tile_f32(gates, kGates, m0 + 16, n0,      lane, acc10);
    store_tile_f32(gates, kGates, m0 + 16, n0 + 16, lane, acc11);
}

// LSTM cell: c' = sig(f)*c + sig(i)*tanh(g); h' = sig(o)*tanh(c')
__global__ void __launch_bounds__(256)
k_lstm(const float* __restrict__ gates, float* __restrict__ c,
       bf16_t* __restrict__ hbf) {
    int idx = blockIdx.x * 256 + threadIdx.x;       // 128*512
    int b = idx >> 9, j = idx & 511;
    const float* g = gates + (size_t)b * kGates;
    float ig = g[j], fg = g[j + 512], gg = g[j + 1024], og = g[j + 1536];
    float cn = sigf(fg) * c[idx] + sigf(ig) * tanhf(gg);
    c[idx] = cn;
    hbf[idx] = (bf16_t)(sigf(og) * tanhf(cn));
}

// logits[128][5056] = h_bf @ W_out^T + b_out   (N zero-padded to 5056)
// 2x2 macro-tile per wave: 632 groups = 4 (M) x 158 (N), 79 blocks x 8 waves.
__global__ void __launch_bounds__(256)
k_gemm_logits(const bf16_t* __restrict__ hbf, const bf16_t* __restrict__ Wout,
              const float* __restrict__ bout, float* __restrict__ logits) {
    int wave = threadIdx.x >> 5, lane = threadIdx.x & 31;
    int grp = blockIdx.x * 8 + wave;
    int m0 = (grp & 3) * 32;
    int n0 = (grp >> 2) * 32;
    int c0 = n0 + (lane & 15), c1 = c0 + 16;
    float bv0 = (c0 < kVocab) ? bout[c0] : 0.0f;
    float bv1 = (c1 < kVocab) ? bout[c1] : 0.0f;
    v8f acc00, acc01, acc10, acc11;
#pragma unroll
    for (int i = 0; i < 8; ++i) {
        acc00[i] = bv0; acc01[i] = bv1; acc10[i] = bv0; acc11[i] = bv1;
    }
    for (int k0 = 0; k0 < kHid; k0 += 32) {
        v16bf a0 = load_a_frag(hbf, kHid, m0,      k0, lane);
        v16bf a1 = load_a_frag(hbf, kHid, m0 + 16, k0, lane);
        v16bf b0 = load_b_frag(Wout, kHid, n0,      k0, lane);
        v16bf b1 = load_b_frag(Wout, kHid, n0 + 16, k0, lane);
        acc00 = WMMA_BF16(a0, b0, acc00);
        acc01 = WMMA_BF16(a0, b1, acc01);
        acc10 = WMMA_BF16(a1, b0, acc10);
        acc11 = WMMA_BF16(a1, b1, acc11);
    }
    store_tile_f32(logits, kNPad, m0,      n0,      lane, acc00);
    store_tile_f32(logits, kNPad, m0,      n0 + 16, lane, acc01);
    store_tile_f32(logits, kNPad, m0 + 16, n0,      lane, acc10);
    store_tile_f32(logits, kNPad, m0 + 16, n0 + 16, lane, acc11);
}

// Softmax((logits + gumbel_t)/T) per batch row; writes fp32 out + bf16 y.
__global__ void __launch_bounds__(256)
k_softmax(float* __restrict__ s, const float* __restrict__ gumbel,
          const float* __restrict__ temp, float* __restrict__ out,
          bf16_t* __restrict__ ybf, int t) {
    int b = blockIdx.x, tid = threadIdx.x;
    float invT = 1.0f / temp[0];
    float* row = s + (size_t)b * kNPad;
    const float* grow = gumbel + ((size_t)t * kBatch + b) * kVocab;
    __shared__ float red[256];

    float mx = -3.0e38f;
    for (int v = tid; v < kVocab; v += 256) {
        float val = (row[v] + grow[v]) * invT;
        row[v] = val;
        mx = fmaxf(mx, val);
    }
    red[tid] = mx; __syncthreads();
    for (int w = 128; w > 0; w >>= 1) {
        if (tid < w) red[tid] = fmaxf(red[tid], red[tid + w]);
        __syncthreads();
    }
    mx = red[0]; __syncthreads();

    float sum = 0.0f;
    for (int v = tid; v < kVocab; v += 256) sum += __expf(row[v] - mx);
    red[tid] = sum; __syncthreads();
    for (int w = 128; w > 0; w >>= 1) {
        if (tid < w) red[tid] += red[tid + w];
        __syncthreads();
    }
    float rs = 1.0f / red[0];

    float* orow  = out + ((size_t)b * kTime + t) * kVocab;
    bf16_t* yrow = ybf + (size_t)b * kVPad;
    for (int v = tid; v < kVocab; v += 256) {
        float y = __expf(row[v] - mx) * rs;
        orow[v] = y;
        yrow[v] = (bf16_t)y;
    }
}

// ---------------------------------------------------------------------------
// One-time prep kernels (run at the start of every launch; deterministic)
// ---------------------------------------------------------------------------

__global__ void k_prep_embT(const float* __restrict__ emb, bf16_t* __restrict__ embT) {
    int idx = blockIdx.x * 256 + threadIdx.x;       // 256 * kVPad
    if (idx >= kEmb * kVPad) return;
    int n = idx / kVPad, k = idx % kVPad;
    float v = (k < kVocab) ? emb[(size_t)k * kEmb + n] : 0.0f;
    embT[idx] = (bf16_t)v;
}

__global__ void k_cvt_bf16(const float* __restrict__ src, bf16_t* __restrict__ dst, int n) {
    int idx = blockIdx.x * 256 + threadIdx.x;
    if (idx < n) dst[idx] = (bf16_t)src[idx];
}

__global__ void k_prep_wout(const float* __restrict__ W, bf16_t* __restrict__ dst) {
    int idx = blockIdx.x * 256 + threadIdx.x;       // kNPad * kHid
    if (idx >= kNPad * kHid) return;
    int r = idx / kHid;
    dst[idx] = (bf16_t)((r < kVocab) ? W[idx] : 0.0f);  // flat idx matches for r<kVocab
}

__global__ void k_prep_bias(const float* __restrict__ bih, const float* __restrict__ bhh,
                            float* __restrict__ bias) {
    int idx = blockIdx.x * 256 + threadIdx.x;
    if (idx < kGates) bias[idx] = bih[idx] + bhh[idx];
}

// h0/c0 = noise @ W_fc_init^T + b_fc_init  (128x128x1024, trivial FLOPs)
__global__ void k_init_state(const float* __restrict__ noise, const float* __restrict__ Wfc,
                             const float* __restrict__ bfc, bf16_t* __restrict__ hbf,
                             float* __restrict__ c) {
    int idx = blockIdx.x * 256 + threadIdx.x;       // 128*1024
    if (idx >= kBatch * 2 * kHid) return;
    int b = idx >> 10, j = idx & 1023;
    const float* nr = noise + (size_t)b * kLat;
    const float* wr = Wfc + (size_t)j * kLat;
    float acc = bfc[j];
    for (int k = 0; k < kLat; ++k) acc += nr[k] * wr[k];
    if (j < kHid) hbf[(size_t)b * kHid + j] = (bf16_t)acc;
    else          c[(size_t)b * kHid + (j - kHid)] = acc;
}

// zero x_bf (t=0 input: emb[pad] row is all-zero) and y_bf (incl. K padding)
__global__ void k_zero_bf(bf16_t* __restrict__ xbf, bf16_t* __restrict__ ybf) {
    int idx = blockIdx.x * 256 + threadIdx.x;
    const int nx = kBatch * kEmb, ny = kBatch * kVPad;
    if (idx < nx) xbf[idx] = (bf16_t)0.0f;
    else if (idx < nx + ny) ybf[idx - nx] = (bf16_t)0.0f;
}

// ---------------------------------------------------------------------------

extern "C" void kernel_launch(void* const* d_in, const int* in_sizes, int n_in,
                              void* d_out, int out_size, void* d_ws, size_t ws_size,
                              hipStream_t stream) {
    const float* noise  = (const float*)d_in[0];
    const float* temp   = (const float*)d_in[1];
    const float* gumbel = (const float*)d_in[2];
    const float* Wfc    = (const float*)d_in[3];
    const float* bfc    = (const float*)d_in[4];
    const float* emb    = (const float*)d_in[5];
    const float* Wih    = (const float*)d_in[6];
    const float* Whh    = (const float*)d_in[7];
    const float* bih    = (const float*)d_in[8];
    const float* bhh    = (const float*)d_in[9];
    const float* Wout   = (const float*)d_in[10];
    const float* bout   = (const float*)d_in[11];
    float* out = (float*)d_out;

    // workspace bump allocator (256B aligned); ~16.3 MB total
    char* ws = (char*)d_ws;
    auto alloc = [&](size_t bytes) -> char* {
        char* p = ws;
        ws += (bytes + 255) & ~(size_t)255;
        return p;
    };
    bf16_t* embT_bf = (bf16_t*)alloc((size_t)kEmb * kVPad * 2);
    bf16_t* Wih_bf  = (bf16_t*)alloc((size_t)kGates * kEmb * 2);
    bf16_t* Whh_bf  = (bf16_t*)alloc((size_t)kGates * kHid * 2);
    bf16_t* Wout_bf = (bf16_t*)alloc((size_t)kNPad * kHid * 2);
    float*  bias    = (float*) alloc((size_t)kGates * 4);
    bf16_t* x_bf    = (bf16_t*)alloc((size_t)kBatch * kEmb * 2);
    bf16_t* h_bf    = (bf16_t*)alloc((size_t)kBatch * kHid * 2);
    float*  c_st    = (float*) alloc((size_t)kBatch * kHid * 4);
    float*  gates   = (float*) alloc((size_t)kBatch * kGates * 4);
    float*  logits  = (float*) alloc((size_t)kBatch * kNPad * 4);
    bf16_t* y_bf    = (bf16_t*)alloc((size_t)kBatch * kVPad * 2);

    // ---- prep (weights -> bf16, bias fold, initial state, zero buffers) ----
    k_prep_embT<<<(kEmb * kVPad + 255) / 256, 256, 0, stream>>>(emb, embT_bf);
    k_cvt_bf16 <<<(kGates * kEmb + 255) / 256, 256, 0, stream>>>(Wih, Wih_bf, kGates * kEmb);
    k_cvt_bf16 <<<(kGates * kHid + 255) / 256, 256, 0, stream>>>(Whh, Whh_bf, kGates * kHid);
    k_prep_wout<<<(kNPad * kHid + 255) / 256, 256, 0, stream>>>(Wout, Wout_bf);
    k_prep_bias<<<(kGates + 255) / 256, 256, 0, stream>>>(bih, bhh, bias);
    k_init_state<<<(kBatch * 2 * kHid + 255) / 256, 256, 0, stream>>>(noise, Wfc, bfc, h_bf, c_st);
    k_zero_bf<<<(kBatch * (kEmb + kVPad) + 255) / 256, 256, 0, stream>>>(x_bf, y_bf);

    // ---- 128 sequential timesteps ----
    for (int t = 0; t < kTime; ++t) {
        if (t > 0) {
            // x = y_{t-1} @ emb   (128 tiles, 8 waves/block)
            k_gemm_x<<<16, 256, 0, stream>>>(y_bf, embT_bf, x_bf);
        }
        // gates = x@Wih^T + h@Whh^T + bias  (256 2x2 macro-tiles)
        k_gemm_gates<<<32, 256, 0, stream>>>(x_bf, h_bf, Wih_bf, Whh_bf, bias, gates);
        // LSTM cell
        k_lstm<<<(kBatch * kHid) / 256, 256, 0, stream>>>(gates, c_st, h_bf);
        // logits = h@Wout^T + b_out  (632 2x2 macro-tiles)
        k_gemm_logits<<<79, 256, 0, stream>>>(h_bf, Wout_bf, bout, logits);
        // y = softmax((logits + g_t)/T); write output row (b,t,:) + next y_bf
        k_softmax<<<kBatch, 256, 0, stream>>>(logits, gumbel, temp, out, y_bf, t);
    }
}
